// RelationalGraphAttentionLayer_4552665334369
// MI455X (gfx1250) — matrix operand order repaired and checked by reading, not compiled, not run on hardware
//
#include <hip/hip_runtime.h>

#define IN_DIMC 256
#define OUT_DIMC 64
#define R_DIMC 64
#define NMREL 4
#define LRELU_SLOPE 0.2f
#define EPS_C 1e-9f

typedef __attribute__((ext_vector_type(16))) __bf16 v16bf;
typedef __attribute__((ext_vector_type(8)))  float  v8f;

// ---------- A fragment (16x32 bf16) from f32 row-major, per CDNA5 ISA §7.12.2 ----------
// lane l: row = l&15, half = l>>4. Lane holds two contiguous 8-float runs:
//   k0 + 8*half + [0..8)  and  k0 + 16 + 8*half + [0..8)
__device__ __forceinline__ v16bf load_a_f32(const float* __restrict__ Arow, int k0, int half) {
  const float4 p0 = *(const float4*)(Arow + k0 + 8 * half);
  const float4 p1 = *(const float4*)(Arow + k0 + 8 * half + 4);
  const float4 p2 = *(const float4*)(Arow + k0 + 16 + 8 * half);
  const float4 p3 = *(const float4*)(Arow + k0 + 16 + 8 * half + 4);
  v16bf a;
  a[0]  = (__bf16)p0.x; a[1]  = (__bf16)p0.y; a[2]  = (__bf16)p0.z; a[3]  = (__bf16)p0.w;
  a[4]  = (__bf16)p1.x; a[5]  = (__bf16)p1.y; a[6]  = (__bf16)p1.z; a[7]  = (__bf16)p1.w;
  a[8]  = (__bf16)p2.x; a[9]  = (__bf16)p2.y; a[10] = (__bf16)p2.z; a[11] = (__bf16)p2.w;
  a[12] = (__bf16)p3.x; a[13] = (__bf16)p3.y; a[14] = (__bf16)p3.z; a[15] = (__bf16)p3.w;
  return a;
}

// ---------- B fragment from pre-swizzled (fragment-major) weights ----------
// Packed layout: frag[(kt*4 + c)*32 + lane] is the v16bf this lane feeds to WMMA.
__device__ __forceinline__ v16bf load_b_packed(const __bf16* __restrict__ B,
                                               int kt, int c, int lane) {
  return ((const v16bf*)B)[((kt * 4 + c) * 32) + lane];
}

// Decompose packed index -> (k, o) of the source weight element.
__device__ __forceinline__ void packed_idx(int ii, int& k, int& o) {
  int j  = ii & 15;         // element within lane fragment
  int l  = (ii >> 4) & 31;  // lane
  int ct = ii >> 9;         // (kt*4 + c)
  int c  = ct & 3;
  int kt = ct >> 2;
  k = kt * 32 + 8 * (l >> 4) + (j & 7) + 16 * (j >> 3);
  o = c * 16 + (l & 15);
}

// ---------- weight conversion: f32 -> bf16, fragment-major ----------
__global__ void k_convert(const float* __restrict__ Wm, const float* __restrict__ Wr,
                          const float* __restrict__ out_w,
                          __bf16* __restrict__ WmP, __bf16* __restrict__ WrP,
                          __bf16* __restrict__ outWP) {
  int i = blockIdx.x * blockDim.x + threadIdx.x;
  const int nWm = NMREL * IN_DIMC * OUT_DIMC;      // 65536 (16384 per m)
  const int nWr = R_DIMC * OUT_DIMC;               // 4096
  const int nOw = 2 * OUT_DIMC * NMREL * OUT_DIMC; // 512*64 = 32768
  if (i < nWm) {
    int m = i >> 14;           // / 16384
    int ii = i & 16383;
    int k, o; packed_idx(ii, k, o);
    WmP[i] = (__bf16)Wm[((long long)m * OUT_DIMC + o) * IN_DIMC + k];
  } else if (i < nWm + nWr) {
    int ii = i - nWm;
    int k, o; packed_idx(ii, k, o);
    WrP[ii] = (__bf16)Wr[o * R_DIMC + k];
  } else if (i < nWm + nWr + nOw) {
    int ii = i - nWm - nWr;
    int k, o; packed_idx(ii, k, o);
    outWP[ii] = (__bf16)out_w[(long long)o * (2 * OUT_DIMC * NMREL) + k];
  }
}

__global__ void k_zero(float* __restrict__ p, long long cnt) {
  long long i = (long long)blockIdx.x * blockDim.x + threadIdx.x;
  long long stride = (long long)gridDim.x * blockDim.x;
  for (; i < cnt; i += stride) p[i] = 0.f;
}

// ---------- Wh = h @ Wm[m]^T  (+ fused per-node attention dots) ----------
// grid: ceil(n/16) blocks of 128 threads; wave w handles relation m = w.
__global__ void k_wh(const float* __restrict__ h, const __bf16* __restrict__ WmP,
                     const float* __restrict__ attn_w,
                     float* __restrict__ Wh, float* __restrict__ s_src,
                     float* __restrict__ s_dst, int n) {
  int m = threadIdx.x >> 5;
  int lane = threadIdx.x & 31;
  int half = lane >> 4, r = lane & 15;
  int row0 = blockIdx.x * 16;
  int arow = row0 + r; if (arow > n - 1) arow = n - 1;
  const float* Arow = h + (long long)arow * IN_DIMC;
  const __bf16* Bm = WmP + (long long)m * IN_DIMC * OUT_DIMC;

  v8f acc[4] = {};
#pragma unroll
  for (int kt = 0; kt < IN_DIMC / 32; ++kt) {
    v16bf a = load_a_f32(Arow, kt * 32, half);
#pragma unroll
    for (int c = 0; c < 4; ++c) {
      v16bf b = load_b_packed(Bm, kt, c, lane);
      acc[c] = __builtin_amdgcn_wmma_f32_16x16x32_bf16(false, a, false, b,
                                                       (short)0, acc[c], false, false);
    }
  }

  float ps[8], pd[8];
#pragma unroll
  for (int v = 0; v < 8; ++v) { ps[v] = 0.f; pd[v] = 0.f; }
#pragma unroll
  for (int c = 0; c < 4; ++c) {
    float as = attn_w[c * 16 + r];
    float ad = attn_w[OUT_DIMC + c * 16 + r];
#pragma unroll
    for (int v = 0; v < 8; ++v) {
      float x = acc[c][v];
      int row = row0 + 8 * half + v;
      if (row < n) Wh[((long long)m * n + row) * OUT_DIMC + c * 16 + r] = x;
      ps[v] += x * as;
      pd[v] += x * ad;
    }
  }
  // reduce across the 16 lanes of each half-group
#pragma unroll
  for (int sh = 1; sh < 16; sh <<= 1) {
#pragma unroll
    for (int v = 0; v < 8; ++v) {
      ps[v] += __shfl_xor(ps[v], sh, 32);
      pd[v] += __shfl_xor(pd[v], sh, 32);
    }
  }
#pragma unroll
  for (int v = 0; v < 8; ++v) {
    if (r == v) {
      int row = row0 + 8 * half + v;
      if (row < n) {
        s_src[(long long)m * n + row] = ps[v];
        s_dst[(long long)m * n + row] = pd[v];
      }
    }
  }
}

// ---------- rel_alpha = exp(relu(r_ij @ Wr^T + bm1) @ Wm2^T + bm2) ----------
// block 256 = 8 waves, each wave one 16-edge tile.
__global__ void k_rel(const float* __restrict__ r_ij, const __bf16* __restrict__ WrP,
                      const float* __restrict__ bm1, const float* __restrict__ Wm2,
                      const float* __restrict__ bm2,
                      float* __restrict__ rel_alpha, int E) {
  int wave = threadIdx.x >> 5;
  int lane = threadIdx.x & 31;
  int half = lane >> 4, r = lane & 15;
  int row0 = (blockIdx.x * 8 + wave) * 16;
  if (row0 >= E) return;
  int arow = row0 + r; if (arow > E - 1) arow = E - 1;
  const float* Arow = r_ij + (long long)arow * R_DIMC;

  v8f acc[4] = {};
#pragma unroll
  for (int kt = 0; kt < R_DIMC / 32; ++kt) {
    v16bf a = load_a_f32(Arow, kt * 32, half);
#pragma unroll
    for (int c = 0; c < 4; ++c) {
      v16bf b = load_b_packed(WrP, kt, c, lane);
      acc[c] = __builtin_amdgcn_wmma_f32_16x16x32_bf16(false, a, false, b,
                                                       (short)0, acc[c], false, false);
    }
  }

  float ps[8];
#pragma unroll
  for (int v = 0; v < 8; ++v) ps[v] = 0.f;
  float b2 = bm2[0];
#pragma unroll
  for (int c = 0; c < 4; ++c) {
    float w = Wm2[c * 16 + r];
    float bias = bm1[c * 16 + r];
#pragma unroll
    for (int v = 0; v < 8; ++v) {
      float x = acc[c][v] + bias;
      x = (x > 0.f) ? x : 0.f;  // relu
      ps[v] += x * w;
    }
  }
#pragma unroll
  for (int sh = 1; sh < 16; sh <<= 1) {
#pragma unroll
    for (int v = 0; v < 8; ++v) ps[v] += __shfl_xor(ps[v], sh, 32);
  }
#pragma unroll
  for (int v = 0; v < 8; ++v) {
    if (r == v) {
      int row = row0 + 8 * half + v;
      if (row < E) rel_alpha[row] = __expf(ps[v] + b2);
    }
  }
}

// ---------- per-edge scalars: denominators + attn_alpha ----------
__global__ void k_edge(const long long* __restrict__ ei, const float* __restrict__ rel_alpha,
                       const float* __restrict__ s_src, const float* __restrict__ s_dst,
                       float* __restrict__ rel_denom, float* __restrict__ attn_alpha,
                       float* __restrict__ attn_denom, int E, int n) {
  int e = blockIdx.x * blockDim.x + threadIdx.x;
  if (e >= E) return;
  int src = (int)ei[e];
  int dst = (int)ei[(long long)E + e];
  atomicAdd(&rel_denom[dst], rel_alpha[e]);
#pragma unroll
  for (int m = 0; m < NMREL; ++m) {
    float s = s_src[(long long)m * n + src] + s_dst[(long long)m * n + dst];
    s = (s > 0.f) ? s : LRELU_SLOPE * s;
    float a = __expf(s);
    attn_alpha[(long long)m * E + e] = a;
    atomicAdd(&attn_denom[(long long)m * n + dst], a);
  }
}

// ---------- aggregation: one wave per (edge, m); lanes cover 64 dims ----------
__global__ void k_agg(const long long* __restrict__ ei, const float* __restrict__ Wh,
                      const float* __restrict__ rel_alpha, const float* __restrict__ rel_denom,
                      const float* __restrict__ attn_alpha, const float* __restrict__ attn_denom,
                      float* __restrict__ rel_agg, float* __restrict__ attn_agg,
                      int E, int n) {
  long long gw = (long long)blockIdx.x * (blockDim.x >> 5) + (threadIdx.x >> 5);
  int lane = threadIdx.x & 31;
  long long e = gw >> 2;
  int m = (int)(gw & 3);
  if (e >= E) return;
  int src = (int)ei[e], dst = (int)ei[(long long)E + e];
  float beta  = rel_alpha[e] / (rel_denom[dst] + EPS_C);
  float alpha = attn_alpha[(long long)m * E + e] / (attn_denom[(long long)m * n + dst] + EPS_C);
  const float* w = Wh + ((long long)m * n + src) * OUT_DIMC;
  float w0 = w[lane], w1 = w[lane + 32];
  float* ra = rel_agg  + ((long long)dst * NMREL + m) * OUT_DIMC;
  float* aa = attn_agg + ((long long)dst * NMREL + m) * OUT_DIMC;
  atomicAdd(&aa[lane],      alpha * w0);
  atomicAdd(&aa[lane + 32], alpha * w1);
  atomicAdd(&ra[lane],      beta * w0);
  atomicAdd(&ra[lane + 32], beta * w1);
}

// ---------- out = relu([attn_agg | rel_agg] @ out_w^T + out_b) ----------
// block 128 = 4 waves; wave c handles output-column tile c (16 cols).
__global__ void k_out(const float* __restrict__ attn_agg, const float* __restrict__ rel_agg,
                      const __bf16* __restrict__ outWP, const float* __restrict__ out_b,
                      float* __restrict__ out, int n) {
  int c = threadIdx.x >> 5;
  int lane = threadIdx.x & 31;
  int half = lane >> 4, r = lane & 15;
  int row0 = blockIdx.x * 16;
  int arow = row0 + r; if (arow > n - 1) arow = n - 1;
  const float* A0 = attn_agg + (long long)arow * (NMREL * OUT_DIMC);
  const float* A1 = rel_agg  + (long long)arow * (NMREL * OUT_DIMC);

  v8f acc = {};
#pragma unroll
  for (int kt = 0; kt < 8; ++kt) {
    v16bf a = load_a_f32(A0, kt * 32, half);
    v16bf b = load_b_packed(outWP, kt, c, lane);
    acc = __builtin_amdgcn_wmma_f32_16x16x32_bf16(false, a, false, b, (short)0, acc, false, false);
  }
#pragma unroll
  for (int kt = 0; kt < 8; ++kt) {
    v16bf a = load_a_f32(A1, kt * 32, half);
    v16bf b = load_b_packed(outWP, 8 + kt, c, lane);
    acc = __builtin_amdgcn_wmma_f32_16x16x32_bf16(false, a, false, b, (short)0, acc, false, false);
  }

  float bias = out_b[c * 16 + r];
#pragma unroll
  for (int v = 0; v < 8; ++v) {
    int row = row0 + 8 * half + v;
    if (row < n) {
      float x = acc[v] + bias;
      out[(long long)row * OUT_DIMC + c * 16 + r] = (x > 0.f) ? x : 0.f;
    }
  }
}

extern "C" void kernel_launch(void* const* d_in, const int* in_sizes, int n_in,
                              void* d_out, int out_size, void* d_ws, size_t ws_size,
                              hipStream_t stream) {
  const float*     h      = (const float*)d_in[0];
  const long long* ei     = (const long long*)d_in[1];
  const float*     r_ij   = (const float*)d_in[2];
  const float*     Wm     = (const float*)d_in[3];
  const float*     Wr     = (const float*)d_in[4];
  const float*     bm1    = (const float*)d_in[5];
  const float*     Wm2    = (const float*)d_in[6];
  const float*     bm2    = (const float*)d_in[7];
  const float*     attn_w = (const float*)d_in[8];
  const float*     out_w  = (const float*)d_in[9];
  const float*     out_b  = (const float*)d_in[10];
  int n = in_sizes[0] / IN_DIMC;
  int E = in_sizes[2] / R_DIMC;
  float* out = (float*)d_out;

  char* ws = (char*)d_ws;
  size_t off = 0;
  auto alloc = [&](size_t bytes) -> void* {
    void* p = ws + off;
    off += (bytes + 255) & ~(size_t)255;
    return p;
  };

  __bf16* WmP    = (__bf16*)alloc((size_t)NMREL * IN_DIMC * OUT_DIMC * 2);
  __bf16* WrP    = (__bf16*)alloc((size_t)R_DIMC * OUT_DIMC * 2);
  __bf16* outWP  = (__bf16*)alloc((size_t)2 * OUT_DIMC * NMREL * OUT_DIMC * 2);
  float*  Wh     = (float*)alloc((size_t)NMREL * n * OUT_DIMC * 4);
  float*  s_src  = (float*)alloc((size_t)NMREL * n * 4);
  float*  s_dst  = (float*)alloc((size_t)NMREL * n * 4);
  float*  relA   = (float*)alloc((size_t)E * 4);
  float*  attnA  = (float*)alloc((size_t)NMREL * E * 4);
  size_t zoff = off;
  float*  relD   = (float*)alloc((size_t)n * 4);
  float*  attnD  = (float*)alloc((size_t)NMREL * n * 4);
  float*  relAgg = (float*)alloc((size_t)n * NMREL * OUT_DIMC * 4);
  float*  attnAgg= (float*)alloc((size_t)n * NMREL * OUT_DIMC * 4);
  long long zcnt = (long long)(off - zoff) / 4;
  float* zstart = (float*)(ws + zoff);

  // 1. weight conversion (fragment-major bf16) + accumulator clear
  k_convert<<<400, 256, 0, stream>>>(Wm, Wr, out_w, WmP, WrP, outWP);
  k_zero<<<2048, 256, 0, stream>>>(zstart, zcnt);

  int ntiles = (n + 15) / 16;
  int etiles = (E + 15) / 16;

  // 2. node GEMM + fused attention dots
  k_wh<<<ntiles, 128, 0, stream>>>(h, WmP, attn_w, Wh, s_src, s_dst, n);

  // 3. relation MLP GEMM -> rel_alpha
  k_rel<<<(etiles + 7) / 8, 256, 0, stream>>>(r_ij, WrP, bm1, Wm2, bm2, relA, E);

  // 4. per-edge denominators + attn_alpha
  k_edge<<<(E + 255) / 256, 256, 0, stream>>>(ei, relA, s_src, s_dst, relD, attnA, attnD, E, n);

  // 5. message aggregation (gather + atomic scatter)
  long long waves = (long long)E * NMREL;
  k_agg<<<(int)((waves + 7) / 8), 256, 0, stream>>>(ei, Wh, relA, relD, attnA, attnD,
                                                    relAgg, attnAgg, E, n);

  // 6. output GEMM + bias + relu
  k_out<<<ntiles, 128, 0, stream>>>(attnAgg, relAgg, outWP, out_b, out, n);
}
